// GRUPredictor_84172769067122
// MI455X (gfx1250) — matrix operand order repaired and checked by reading, not compiled, not run on hardware
//
#include <hip/hip_runtime.h>

// ---------------------------------------------------------------------------
// GRU predictor for MI455X (gfx1250, wave32, WMMA), persistent kernel with
// LDS-resident weights.  544 dependent steps in ONE launch with a device-wide
// atomic barrier; recurrent weights are staged once into LDS (pre-swizzled
// into WMMA B-operand order, 48B slots for conflict-free ds_load_b128) and
// re-read from LDS every step instead of pulling ~48 MB/step from L2.
// ---------------------------------------------------------------------------

#define HID   512
#define INP   128
#define OUTD  128
#define BATCH 512
#define SEQ   512
#define PRED  32
#define NBLK  128      // 128 blocks x 256 threads = 1024 waves = 32x32 tiles

#define SLOT_B     48                       // 32B payload + 16B bank pad
#define NSLOT_WHH  (3 * 16 * 32)            // gates x ksteps x lanes = 1536
#define NSLOT_WIH  (3 * 4 * 32)             // gates x ksteps x lanes = 384
#define SMEM_WHH_B (NSLOT_WHH * SLOT_B)     // 73728 B
#define SMEM_BYTES (SMEM_WHH_B + NSLOT_WIH * SLOT_B)  // 92160 B (< 320KB LDS)

typedef __attribute__((ext_vector_type(16))) __bf16 v16bf;
typedef __attribute__((ext_vector_type(8)))  __bf16 v8bf;
typedef __attribute__((ext_vector_type(8)))  float  v8f;
typedef __attribute__((ext_vector_type(4)))  float  v4f;

static __device__ __forceinline__ v8f wmma_bf16(v16bf a, v16bf b, v8f c) {
  // 8 args: (neg_a, A, neg_b, B, c_mod, C, reuse_a, reuse_b)
  return __builtin_amdgcn_wmma_f32_16x16x32_bf16(
      false, a, false, b, (short)0, c, false, false);
}

// A operand (16x32 bf16, M x K): lane L<16 holds row M=L, K in {kc..kc+15}
// with kc=0; lanes 16..31 same rows with kc=16.
static __device__ __forceinline__ v16bf load_a_bf16(const __bf16* row, int kc) {
  v8bf lo = *(const v8bf*)(row + kc);
  v8bf hi = *(const v8bf*)(row + kc + 8);
  v16bf a;
#pragma unroll
  for (int i = 0; i < 8; ++i) { a[i] = lo[i]; a[i + 8] = hi[i]; }
  return a;
}

// Same A layout but sourced from f32 (z_seq) with on-the-fly cvt to bf16.
static __device__ __forceinline__ v16bf load_a_f32(const float* row, int kc) {
  v4f f0 = *(const v4f*)(row + kc);
  v4f f1 = *(const v4f*)(row + kc + 4);
  v4f f2 = *(const v4f*)(row + kc + 8);
  v4f f3 = *(const v4f*)(row + kc + 12);
  v16bf a;
#pragma unroll
  for (int i = 0; i < 4; ++i) {
    a[i]      = (__bf16)f0[i];
    a[4 + i]  = (__bf16)f1[i];
    a[8 + i]  = (__bf16)f2[i];
    a[12 + i] = (__bf16)f3[i];
  }
  return a;
}

// B operand from an LDS slot (32B payload, two ds_load_b128).
static __device__ __forceinline__ v16bf load_b_lds(const char* sm, int slot) {
  const __bf16* p = (const __bf16*)(sm + (size_t)slot * SLOT_B);
  v8bf lo = *(const v8bf*)(p);
  v8bf hi = *(const v8bf*)(p + 8);
  v16bf b;
#pragma unroll
  for (int i = 0; i < 8; ++i) { b[i] = lo[i]; b[i + 8] = hi[i]; }
  return b;
}

static __device__ __forceinline__ float sigmoidf_(float x) {
  return 1.0f / (1.0f + __expf(-x));
}
static __device__ __forceinline__ float tanhf_(float x) {
  x = fminf(fmaxf(x, -15.0f), 15.0f);
  float e = __expf(2.0f * x);
  return (e - 1.0f) / (e + 1.0f);
}

// Device-wide sense-reversal barrier.  bar[0]=arrival counter, bar[1]=gen.
static __device__ __forceinline__ void grid_barrier(unsigned* bar) {
  __threadfence();
  __syncthreads();
  if (threadIdx.x == 0) {
    unsigned g = __hip_atomic_load(&bar[1], __ATOMIC_RELAXED,
                                   __HIP_MEMORY_SCOPE_AGENT);
    unsigned a = __hip_atomic_fetch_add(&bar[0], 1u, __ATOMIC_ACQ_REL,
                                        __HIP_MEMORY_SCOPE_AGENT);
    if (a == (unsigned)(NBLK - 1)) {
      __hip_atomic_store(&bar[0], 0u, __ATOMIC_RELAXED,
                         __HIP_MEMORY_SCOPE_AGENT);
      __hip_atomic_store(&bar[1], g + 1u, __ATOMIC_RELEASE,
                         __HIP_MEMORY_SCOPE_AGENT);
    } else {
      while (__hip_atomic_load(&bar[1], __ATOMIC_ACQUIRE,
                               __HIP_MEMORY_SCOPE_AGENT) == g) {
        __builtin_amdgcn_s_sleep(2);
      }
    }
  }
  __syncthreads();
  __threadfence();
}

// ---------------------------------------------------------------------------
// f32 -> bf16 conversion (weights)
// ---------------------------------------------------------------------------
__global__ __launch_bounds__(256)
void cvt_f32_to_bf16_kernel(const float* __restrict__ src,
                            __bf16* __restrict__ dst, int n) {
  int i = blockIdx.x * 256 + threadIdx.x;
  if (i < n) dst[i] = (__bf16)src[i];
}

// ---------------------------------------------------------------------------
// Persistent GRU kernel: 512 encode + 32 decode steps.
// Tile map: block b owns tile_n = b>>2 (16 hidden units -> its B operands are
// identical for all 8 waves and live in LDS); tile_m = (b&3)*8 + wave.
// LDS slot id for (gate g, kstep t, lane l): Whh (g*16+t)*32+l;
//                                            Wih NSLOT_WHH-rel (g*4+t)*32+l.
// Decode: blocks b<32 also emit output tiles (o-tile = b>>2, m-tile per wave).
// ---------------------------------------------------------------------------
__global__ __launch_bounds__(256)
void gru_persistent_kernel(const float*  __restrict__ z_seq,  // [B, SEQ, INP]
                           __bf16* __restrict__ hb0, float* __restrict__ hf0,
                           __bf16* __restrict__ hb1, float* __restrict__ hf1,
                           const __bf16* __restrict__ Whh,    // [3H, H]
                           const __bf16* __restrict__ Wih,    // [3H, INP]
                           const float*  __restrict__ b_ih,   // [3H]
                           const float*  __restrict__ b_hh,   // [3H]
                           const __bf16* __restrict__ Wout,   // [OUTD, H]
                           const float*  __restrict__ b_out,  // [OUTD]
                           float*        __restrict__ out,    // [B, PRED, OUTD]
                           unsigned*     __restrict__ bar)
{
  extern __shared__ char smem[];                 // 92160 B
  char* smU = smem + SMEM_WHH_B;

  const int lane = threadIdx.x & 31;
  const int wave = threadIdx.x >> 5;
  const int blk  = blockIdx.x;

  const int tile_n = blk >> 2;                   // 0..31 (shared by block)
  const int tile_m = ((blk & 3) << 3) + wave;    // 0..31 (per wave)
  const int m0 = tile_m * 16;
  const int j0 = tile_n * 16;

  const int arow = m0 + (lane & 15);   // batch row for A operand
  const int kc   = (lane >> 4) << 4;   // A K-chunk base: 0 or 16
  const int koff = (lane >> 4) << 4;   // B K-chunk base: 0 or 16
  const int j    = j0 + (lane & 15);   // hidden unit (weight row)
  const int mhi  = (lane >> 4) << 3;

  // ---- One-time fill: stage this block's weight slice into LDS, already
  // swizzled into B-operand lane order.  48B slots -> stride-12-DWORD bank
  // map, only the minimal 2-way (L vs L+16) sharing per ds_load_b128.
  for (int sid = threadIdx.x; sid < NSLOT_WHH; sid += 256) {
    const int g = sid >> 9, t = (sid >> 5) & 15, l = sid & 31;
    const int jj = j0 + (l & 15), ko = (l >> 4) << 4;
    const __bf16* src = Whh + ((size_t)(g * HID + jj)) * HID + t * 32 + ko;
    __bf16* dst = (__bf16*)(smem + (size_t)sid * SLOT_B);
    *(v8bf*)(dst)     = *(const v8bf*)(src);
    *(v8bf*)(dst + 8) = *(const v8bf*)(src + 8);
  }
  for (int sid = threadIdx.x; sid < NSLOT_WIH; sid += 256) {
    const int g = sid >> 7, t = (sid >> 5) & 3, l = sid & 31;
    const int jj = j0 + (l & 15), ko = (l >> 4) << 4;
    const __bf16* src = Wih + ((size_t)(g * HID + jj)) * INP + t * 32 + ko;
    __bf16* dst = (__bf16*)(smU + (size_t)sid * SLOT_B);
    *(v8bf*)(dst)     = *(const v8bf*)(src);
    *(v8bf*)(dst + 8) = *(const v8bf*)(src + 8);
  }
  __syncthreads();

  // Per-lane bias terms (loop-invariant).
  const float br_b  = b_ih[j]       + b_hh[j];
  const float bz_b  = b_ih[HID + j] + b_hh[HID + j];
  const float bhn_b = b_hh[2 * HID + j];
  const float bin_b = b_ih[2 * HID + j];

  for (int s = 0; s < SEQ + PRED; ++s) {
    const int enc = (s < SEQ);
    const __bf16* hbi = (s & 1) ? hb1 : hb0;
    const float*  hfi = (s & 1) ? hf1 : hf0;
    __bf16*       hbo = (s & 1) ? hb0 : hb1;
    float*        hfo = (s & 1) ? hf0 : hf1;

    v8f c_r = {}, c_z = {}, c_hn = {}, c_in = {};
    const __bf16* ap = hbi + (size_t)arow * HID;

#pragma unroll 4
    for (int t = 0; t < 16; ++t) {       // K = 512 over hidden state
      v16bf a = load_a_bf16(ap + t * 32, kc);
      c_r  = wmma_bf16(a, load_b_lds(smem, (0 * 16 + t) * 32 + lane), c_r);
      c_z  = wmma_bf16(a, load_b_lds(smem, (1 * 16 + t) * 32 + lane), c_z);
      c_hn = wmma_bf16(a, load_b_lds(smem, (2 * 16 + t) * 32 + lane), c_hn);
    }

    if (enc) {                           // K = 128 over z_t (HBM stream)
      const float* zr = z_seq + ((size_t)arow * SEQ + s) * INP;
      __builtin_prefetch(zr + INP, 0, 3);   // next step's z row
#pragma unroll
      for (int t = 0; t < 4; ++t) {
        v16bf a = load_a_f32(zr + t * 32, kc);
        c_r  = wmma_bf16(a, load_b_lds(smU, (0 * 4 + t) * 32 + lane), c_r);
        c_z  = wmma_bf16(a, load_b_lds(smU, (1 * 4 + t) * 32 + lane), c_z);
        c_in = wmma_bf16(a, load_b_lds(smU, (2 * 4 + t) * 32 + lane), c_in);
      }
    }

    // Gate math (f32).  D layout: VGPR r -> M = r + 8*(lane>=16), N = lane&15.
#pragma unroll
    for (int r = 0; r < 8; ++r) {
      const int b = m0 + r + mhi;
      float rg = sigmoidf_(c_r[r] + br_b);
      float zg = sigmoidf_(c_z[r] + bz_b);
      float hn = c_hn[r] + bhn_b;
      float in = (enc ? c_in[r] : 0.0f) + bin_b;
      float n  = tanhf_(in + rg * hn);
      float hp = hfi[(size_t)b * HID + j];
      float hv = (1.0f - zg) * n + zg * hp;
      hfo[(size_t)b * HID + j] = hv;
      hbo[(size_t)b * HID + j] = (__bf16)hv;
    }

    grid_barrier(bar);  // publish h_new device-wide

    // Decode output projection (reads the buffer just published; disjoint
    // from the buffers written by step s+1).  Blocks 0..31 cover 8x32 tiles.
    if (!enc && blk < 32) {
      const int d  = s - SEQ;
      const int o0 = (blk >> 2) * 16;
      const int o  = o0 + (lane & 15);
      v8f c = {};
      const __bf16* hap = hbo + (size_t)arow * HID;
      const __bf16* wp  = Wout + (size_t)o * HID;
#pragma unroll 4
      for (int kt = 0; kt < HID; kt += 32) {
        v16bf a = load_a_bf16(hap + kt, kc);
        c = wmma_bf16(a, *(const v16bf*)(wp + kt + koff), c);
      }
      const float bo = b_out[o];
#pragma unroll
      for (int r = 0; r < 8; ++r) {
        const int b = m0 + r + mhi;
        out[((size_t)b * PRED + d) * OUTD + o] = c[r] + bo;
      }
    }
  }
}

// ---------------------------------------------------------------------------
// Host launch
// ---------------------------------------------------------------------------
extern "C" void kernel_launch(void* const* d_in, const int* in_sizes, int n_in,
                              void* d_out, int out_size, void* d_ws, size_t ws_size,
                              hipStream_t stream) {
  (void)in_sizes; (void)n_in; (void)out_size; (void)ws_size;

  const float* z_seq = (const float*)d_in[0];  // [B, L, INP]
  const float* W_ih  = (const float*)d_in[1];  // [3H, INP]
  const float* W_hh  = (const float*)d_in[2];  // [3H, H]
  const float* b_ih  = (const float*)d_in[3];  // [3H]
  const float* b_hh  = (const float*)d_in[4];  // [3H]
  const float* W_out = (const float*)d_in[5];  // [OUTD, H]
  const float* b_out = (const float*)d_in[6];  // [OUTD]
  float* out = (float*)d_out;                  // [B, PRED, OUTD]

  // Workspace layout (256B-aligned carves), total ~5.3 MB.
  char* ws = (char*)d_ws;
  size_t off = 0;
  auto carve = [&](size_t bytes) {
    void* p = ws + off;
    off = (off + bytes + 255) & ~(size_t)255;
    return p;
  };
  const int nWhh = 3 * HID * HID;     // 786432
  const int nWih = 3 * HID * INP;     // 196608
  const int nWo  = OUTD * HID;        // 65536
  const size_t nH = (size_t)BATCH * HID;

  __bf16*   Whh_b  = (__bf16*)carve((size_t)nWhh * 2);
  __bf16*   Wih_b  = (__bf16*)carve((size_t)nWih * 2);
  __bf16*   Wout_b = (__bf16*)carve((size_t)nWo  * 2);
  float*    hf0 = (float*) carve(nH * 4);
  float*    hf1 = (float*) carve(nH * 4);
  __bf16*   hb0 = (__bf16*)carve(nH * 2);
  __bf16*   hb1 = (__bf16*)carve(nH * 2);
  unsigned* bar = (unsigned*)carve(256);

  // 1) Convert weights to bf16 (every call; deterministic).
  cvt_f32_to_bf16_kernel<<<(nWhh + 255) / 256, 256, 0, stream>>>(W_hh,  Whh_b,  nWhh);
  cvt_f32_to_bf16_kernel<<<(nWih + 255) / 256, 256, 0, stream>>>(W_ih,  Wih_b,  nWih);
  cvt_f32_to_bf16_kernel<<<(nWo  + 255) / 256, 256, 0, stream>>>(W_out, Wout_b, nWo);

  // 2) h0 = 0; barrier state = 0.
  hipMemsetAsync(hf0, 0, nH * 4, stream);
  hipMemsetAsync(hb0, 0, nH * 2, stream);
  hipMemsetAsync(bar, 0, 256, stream);

  // 3) One persistent launch runs all 544 steps + decode projections.
  gru_persistent_kernel<<<NBLK, 256, SMEM_BYTES, stream>>>(
      z_seq, hb0, hf0, hb1, hf1, Whh_b, Wih_b, b_ih, b_hh,
      Wout_b, b_out, out, bar);
}
